// SGVAE_58944131170770
// MI455X (gfx1250) — matrix operand order repaired and checked by reading, not compiled, not run on hardware
//
#include <hip/hip_runtime.h>

typedef __bf16 bf16;
typedef __attribute__((ext_vector_type(16))) __bf16 v16bf;
typedef __attribute__((ext_vector_type(8)))  float  v8f;
typedef __attribute__((ext_vector_type(4)))  int    v4i;

#define B_DIM  8192
#define D_IN   4096
#define H_DIM  4096
#define L_DIM  1024

// --- CDNA5 async global->LDS copy (guarded; falls back to sync copy) -------
#if defined(__has_builtin)
#if __has_builtin(__builtin_amdgcn_global_load_async_to_lds_b128)
#define HAVE_ASYNC_LDS 1
#endif
#endif

#define AS1 __attribute__((address_space(1)))
#define AS3 __attribute__((address_space(3)))

__device__ __forceinline__ void wait_async_zero() {
#if defined(__has_builtin)
#if __has_builtin(__builtin_amdgcn_s_wait_asynccnt)
  __builtin_amdgcn_s_wait_asynccnt(0);
  return;
#else
  asm volatile("s_wait_asynccnt 0x0" ::: "memory");
  return;
#endif
#endif
}

// ---------------------------------------------------------------------------
// f32 -> bf16 conversion, 4 elems/thread (n multiple of 4 — true here)
// ---------------------------------------------------------------------------
__global__ void k_f32_to_bf16(const float* __restrict__ in, bf16* __restrict__ out, long n) {
  long i = ((long)blockIdx.x * blockDim.x + threadIdx.x) * 4;
  long stride = (long)gridDim.x * blockDim.x * 4;
  for (; i < n; i += stride) {
    float4 f = *(const float4*)(in + i);
    union { bf16 h[4]; unsigned long long u; } p;
    p.h[0] = (bf16)f.x; p.h[1] = (bf16)f.y; p.h[2] = (bf16)f.z; p.h[3] = (bf16)f.w;
    *(unsigned long long*)(out + i) = p.u;
  }
}

// ---------------------------------------------------------------------------
// WMMA bf16 GEMM: C[M,N] = act(A[M,K] @ B[K,N] + bias[N])
// Block tile 128x128, k-step 32, 256 threads = 8 waves (4x2 wave grid);
// wave tile 32x64 = 2x4 v_wmma_f32_16x16x32_bf16 per k-step.
// Double-buffered LDS pipeline: stage tile k+1 (A via async global->LDS when
// available) while computing tile k; ONE barrier per k-step.
// A staged row-major; B staged transposed (sBT[n][k]) -> all fragments are
// contiguous ds_load_b128 pairs. Dims are multiples of tile sizes here.
// ---------------------------------------------------------------------------
template<bool RELU, typename OutT>
__global__ __launch_bounds__(256) void k_gemm_bf16(
    const bf16* __restrict__ A, const bf16* __restrict__ Bm,
    const float* __restrict__ bias, OutT* __restrict__ C,
    int M, int N, int K) {
  const int BM = 128, BN = 128, BK = 32;
  const int LDA = BK + 8;   // 40 elems -> 80B pitch (16B aligned)
  const int LDT = BK + 8;
  __shared__ bf16 sA [2][128 * 40];   // [buf][m][k]
  __shared__ bf16 sBT[2][128 * 40];   // [buf][n][k] (transposed B tile)

  const int tid  = threadIdx.x;
  const int lane = tid & 31;
  const int w    = tid >> 5;
  const int wm   = w >> 1;        // 0..3  -> M offset wm*32
  const int wn   = w & 1;         // 0..1  -> N offset wn*64
  const int bM   = blockIdx.y * BM;
  const int bN   = blockIdx.x * BN;
  const bool hiHalf = lane >= 16;
  const int  l16    = lane & 15;

  v8f acc[2][4];
  #pragma unroll
  for (int t = 0; t < 2; ++t)
    #pragma unroll
    for (int u = 0; u < 4; ++u)
      #pragma unroll
      for (int e = 0; e < 8; ++e) acc[t][u][e] = 0.0f;

  auto stage = [&](int buf, int k0) {
    // ---- A tile 128x32: 512 x 16B, 2 per thread ----
    #pragma unroll
    for (int it = 0; it < 2; ++it) {
      int v   = tid + it * 256;
      int row = v >> 2;              // 0..127
      int col = (v & 3) << 3;        // 0,8,16,24
      const bf16* gp = A + (size_t)(bM + row) * K + k0 + col;
      bf16* lp = &sA[buf][row * LDA + col];
#if defined(HAVE_ASYNC_LDS)
      __builtin_amdgcn_global_load_async_to_lds_b128(
          (AS1 v4i*)gp, (AS3 v4i*)lp, 0, 0);
#else
      *(uint4*)lp = *(const uint4*)gp;
#endif
    }
    // ---- B tile 32x128 transposed: 512 x 16B, 2 per thread,
    //      scattered as 8 b16 stores into sBT[n][k] ----
    #pragma unroll
    for (int it = 0; it < 2; ++it) {
      int v   = tid + it * 256;      // 0..511
      int row = v >> 4;              // k row 0..31
      int cv  = (v & 15) << 3;       // col 0..120 step 8
      uint4 d = *(const uint4*)(Bm + (size_t)(k0 + row) * N + bN + cv);
      const bf16* dp = (const bf16*)&d;
      #pragma unroll
      for (int j = 0; j < 8; ++j) sBT[buf][(cv + j) * LDT + row] = dp[j];
    }
  };

  auto compute = [&](int buf) {
    // A fragments (ISA 16-bit A 16x32 layout): VGPRs 0-3 = K kb..kb+7,
    // VGPRs 4-7 = K kb+16..kb+23, kb = hiHalf?8:0
    union { v16bf v; uint4 q[2]; } fa[2];
    #pragma unroll
    for (int t = 0; t < 2; ++t) {
      int row = wm * 32 + t * 16 + l16;
      int kb  = hiHalf ? 8 : 0;
      fa[t].q[0] = *(const uint4*)&sA[buf][row * LDA + kb];
      fa[t].q[1] = *(const uint4*)&sA[buf][row * LDA + 16 + kb];
    }
    // B fragments: lane=col; VGPRs 0-3 = K kb..kb+7, VGPRs 4-7 = K kb+16..kb+23, kb = hiHalf?8:0
    union { v16bf v; uint4 q[2]; } fb[4];
    #pragma unroll
    for (int u = 0; u < 4; ++u) {
      int col = wn * 64 + u * 16 + l16;
      int kb  = hiHalf ? 8 : 0;
      fb[u].q[0] = *(const uint4*)&sBT[buf][col * LDT + kb];
      fb[u].q[1] = *(const uint4*)&sBT[buf][col * LDT + 16 + kb];
    }
    #pragma unroll
    for (int t = 0; t < 2; ++t)
      #pragma unroll
      for (int u = 0; u < 4; ++u)
        acc[t][u] = __builtin_amdgcn_wmma_f32_16x16x32_bf16(
            false, fa[t].v, false, fb[u].v, (short)0, acc[t][u], false, false);
  };

  const int nk = K / BK;
  stage(0, 0);
#if defined(HAVE_ASYNC_LDS)
  wait_async_zero();
#endif
  __syncthreads();
  for (int ki = 1; ki < nk; ++ki) {
    stage(ki & 1, ki * BK);        // fill next buffer
    compute((ki - 1) & 1);         // math on current buffer
#if defined(HAVE_ASYNC_LDS)
    wait_async_zero();
#endif
    __syncthreads();
  }
  compute((nk - 1) & 1);

  // ---- epilogue: C/D layout -> VGPR r holds M = r (+8 for lanes>=16), N = lane&15
  #pragma unroll
  for (int t = 0; t < 2; ++t) {
    #pragma unroll
    for (int u = 0; u < 4; ++u) {
      int col = bN + wn * 64 + u * 16 + l16;
      float bv = bias[col];
      #pragma unroll
      for (int r = 0; r < 8; ++r) {
        int row = bM + wm * 32 + t * 16 + r + (hiHalf ? 8 : 0);
        float vv = acc[t][u][r] + bv;
        if (RELU) vv = fmaxf(vv, 0.0f);
        C[(size_t)row * N + col] = (OutT)vv;
      }
    }
  }
}

// ---------------------------------------------------------------------------
// z = mu + exp(logvar_e)*eps (bf16 out for next GEMM) + KL partial sum
// ---------------------------------------------------------------------------
__global__ __launch_bounds__(256) void k_z_kl(
    const float* __restrict__ enc, const float* __restrict__ eps,
    bf16* __restrict__ z, float* __restrict__ acc_kl) {
  __shared__ float red[256];
  const size_t n = (size_t)B_DIM * L_DIM;
  size_t i = (size_t)blockIdx.x * blockDim.x + threadIdx.x;
  const size_t stride = (size_t)gridDim.x * blockDim.x;
  float local = 0.0f;
  for (; i < n; i += stride) {
    size_t row = i >> 10, col = i & (L_DIM - 1);
    const float* e = enc + row * (2 * L_DIM);
    float mu = e[col];
    float lv = e[L_DIM + col];
    float sd = expf(lv);                 // std = exp(logvar) per reference
    float zz = mu + sd * eps[i];
    z[i] = (bf16)zz;
    local += 0.5f * (sd * sd + mu * mu - 1.0f - 2.0f * lv);
  }
  red[threadIdx.x] = local;
  __syncthreads();
  for (int s = 128; s > 0; s >>= 1) {
    if (threadIdx.x < s) red[threadIdx.x] += red[threadIdx.x + s];
    __syncthreads();
  }
  if (threadIdx.x == 0) atomicAdd(acc_kl, red[0]);
}

// ---------------------------------------------------------------------------
// gaussian NLL: dec layout [B, 2*D_IN]: rmu = dec[:, :D], logvar_d = dec[:, D:]
// ---------------------------------------------------------------------------
__global__ __launch_bounds__(256) void k_recon(
    const float* __restrict__ dec, const float* __restrict__ x,
    float* __restrict__ acc_recon) {
  __shared__ float red[256];
  const size_t n = (size_t)B_DIM * D_IN;
  size_t i = (size_t)blockIdx.x * blockDim.x + threadIdx.x;
  const size_t stride = (size_t)gridDim.x * blockDim.x;
  float local = 0.0f;
  for (; i < n; i += stride) {
    size_t row = i >> 12, col = i & (D_IN - 1);
    const float* d = dec + row * (2 * D_IN);
    float rmu = d[col];
    float lv  = d[D_IN + col];
    float var = expf(2.0f * lv);
    float vc  = fmaxf(var, 1e-6f);
    float df  = x[i] - rmu;
    local += 0.5f * (logf(vc) + df * df / vc);
  }
  red[threadIdx.x] = local;
  __syncthreads();
  for (int s = 128; s > 0; s >>= 1) {
    if (threadIdx.x < s) red[threadIdx.x] += red[threadIdx.x + s];
    __syncthreads();
  }
  if (threadIdx.x == 0) atomicAdd(acc_recon, red[0]);
}

__global__ void k_init_acc(float* acc) {
  if (threadIdx.x < 2) acc[threadIdx.x] = 0.0f;
}

__global__ void k_finalize(const float* __restrict__ acc, float* __restrict__ out) {
  out[0] = (acc[0] + acc[1]) * (1.0f / (float)B_DIM);
}

// ---------------------------------------------------------------------------
extern "C" void kernel_launch(void* const* d_in, const int* in_sizes, int n_in,
                              void* d_out, int out_size, void* d_ws, size_t ws_size,
                              hipStream_t stream) {
  const float* x   = (const float*)d_in[0];
  const float* eps = (const float*)d_in[1];
  const float* We1 = (const float*)d_in[2];
  const float* be1 = (const float*)d_in[3];
  const float* We2 = (const float*)d_in[4];
  const float* be2 = (const float*)d_in[5];
  const float* Wd1 = (const float*)d_in[6];
  const float* bd1 = (const float*)d_in[7];
  const float* Wd2 = (const float*)d_in[8];
  const float* bd2 = (const float*)d_in[9];
  float* out = (float*)d_out;

  char* ws = (char*)d_ws;
  size_t off = 0;
  float* acc = (float*)(ws + off);                 off += 256;
  bf16* xb   = (bf16*)(ws + off);                  off += (size_t)B_DIM * D_IN * 2;
  bf16* we1b = (bf16*)(ws + off);                  off += (size_t)D_IN * H_DIM * 2;
  bf16* we2b = (bf16*)(ws + off);                  off += (size_t)H_DIM * 2 * L_DIM * 2;
  bf16* wd1b = (bf16*)(ws + off);                  off += (size_t)L_DIM * H_DIM * 2;
  bf16* wd2b = (bf16*)(ws + off);                  off += (size_t)H_DIM * 2 * D_IN * 2;
  bf16* hb   = (bf16*)(ws + off);                  off += (size_t)B_DIM * H_DIM * 2;
  float* enc = (float*)(ws + off);                 off += (size_t)B_DIM * 2 * L_DIM * 4;
  bf16* zb   = (bf16*)(ws + off);                  off += (size_t)B_DIM * L_DIM * 2;
  bf16* hdb  = (bf16*)(ws + off);                  off += (size_t)B_DIM * H_DIM * 2;
  float* dec = (float*)(ws + off);                 off += (size_t)B_DIM * 2 * D_IN * 4;

  k_init_acc<<<1, 32, 0, stream>>>(acc);

  // conversions (4 elems per thread)
  k_f32_to_bf16<<<1024, 256, 0, stream>>>(x,   xb,   (long)B_DIM * D_IN);
  k_f32_to_bf16<<<1024, 256, 0, stream>>>(We1, we1b, (long)D_IN * H_DIM);
  k_f32_to_bf16<<<512,  256, 0, stream>>>(We2, we2b, (long)H_DIM * 2 * L_DIM);
  k_f32_to_bf16<<<256,  256, 0, stream>>>(Wd1, wd1b, (long)L_DIM * H_DIM);
  k_f32_to_bf16<<<1024, 256, 0, stream>>>(Wd2, wd2b, (long)H_DIM * 2 * D_IN);

  // encoder GEMM1: h = relu(x @ We1 + be1)   [8192 x 4096]
  k_gemm_bf16<true, bf16><<<dim3(H_DIM / 128, B_DIM / 128), 256, 0, stream>>>(
      xb, we1b, be1, hb, B_DIM, H_DIM, D_IN);

  // encoder GEMM2: enc = h @ We2 + be2       [8192 x 2048] f32
  k_gemm_bf16<false, float><<<dim3((2 * L_DIM) / 128, B_DIM / 128), 256, 0, stream>>>(
      hb, we2b, be2, enc, B_DIM, 2 * L_DIM, H_DIM);

  // reparameterize + KL
  k_z_kl<<<2048, 256, 0, stream>>>(enc, eps, zb, acc + 1);

  // decoder GEMM3: hd = relu(z @ Wd1 + bd1)  [8192 x 4096]
  k_gemm_bf16<true, bf16><<<dim3(H_DIM / 128, B_DIM / 128), 256, 0, stream>>>(
      zb, wd1b, bd1, hdb, B_DIM, H_DIM, L_DIM);

  // decoder GEMM4: dec = hd @ Wd2 + bd2      [8192 x 8192] f32
  k_gemm_bf16<false, float><<<dim3((2 * D_IN) / 128, B_DIM / 128), 256, 0, stream>>>(
      hdb, wd2b, bd2, dec, B_DIM, 2 * D_IN, H_DIM);

  // reconstruction NLL
  k_recon<<<4096, 256, 0, stream>>>(dec, x, acc + 0);

  // loss = (recon_sum + kl_sum) / B
  k_finalize<<<1, 1, 0, stream>>>(acc, out);
}
